// Gate_11527692222468
// MI455X (gfx1250) — matrix operand order repaired and checked by reading, not compiled, not run on hardware
//
#include <hip/hip_runtime.h>
#include <hip/hip_bf16.h>

typedef __bf16 v4bf  __attribute__((ext_vector_type(4)));
typedef __bf16 v8bf  __attribute__((ext_vector_type(8)));
typedef __bf16 v16bf __attribute__((ext_vector_type(16)));
typedef float  v4f   __attribute__((ext_vector_type(4)));
typedef float  v8f   __attribute__((ext_vector_type(8)));

#define DIM     2048
#define NEXP    256
#define TOPK    8
#define MTILE   64      // tokens per block
#define KCH     128     // K chunk staged in LDS
#define LSTRIDE 136     // padded bf16 row stride (kills 256B-period bank conflicts)
#define LOGSTR  257     // padded f32 logits row stride

#define XS_BYTES (MTILE * LSTRIDE * 2)          // 17408
#define WS_BYTES (NEXP  * LSTRIDE * 2)          // 69632
#define SMEM_BYTES (XS_BYTES + WS_BYTES)        // 87040 (> epilogue's 66816)

// ---- pre-pass: W fp32 -> bf16 (2MB -> 1MB, negligible vs 134MB x stream) ----
__global__ __launch_bounds__(256) void cvt_w_kernel(const float* __restrict__ W,
                                                    __bf16* __restrict__ Wb)
{
    const int i = (blockIdx.x * 256 + threadIdx.x) * 4;
    v4f f = *(const v4f*)(W + i);
    v4bf h;
    h[0] = (__bf16)f[0]; h[1] = (__bf16)f[1];
    h[2] = (__bf16)f[2]; h[3] = (__bf16)f[3];
    *(v4bf*)(Wb + i) = h;
}

__global__ __launch_bounds__(256) void gate_topk_kernel(
    const float* __restrict__ X, const __bf16* __restrict__ Wb,
    const float* __restrict__ B, float* __restrict__ outW,
    int* __restrict__ outI)
{
    __shared__ __align__(16) char smem[SMEM_BYTES];
    __bf16* Xs = (__bf16*)smem;
    __bf16* Ws = (__bf16*)(smem + XS_BYTES);
    float*  Lg = (float*)smem;                        // epilogue alias
    float*  Bs = (float*)(smem + MTILE * LOGSTR * 4); // epilogue alias

    const int tid    = threadIdx.x;
    const int lane   = tid & 31;
    const int wave   = tid >> 5;
    const int tgroup = wave & 3;   // which 16-token group
    const int ehalf  = wave >> 2;  // experts 0-127 or 128-255
    const int t0     = blockIdx.x * MTILE;

    v8f acc[8] = {};

    const int xrow = tid & 63;
    const int xseg = tid >> 6;

    // Per-lane addresses for async W staging:
    //   LDS[lds_off + c*16]  <-  GLOBAL[Wb + k0 + voff + c*16]
    const unsigned lds_woff =
        (unsigned)(uintptr_t)(smem + XS_BYTES) + (unsigned)tid * (LSTRIDE * 2);
    const unsigned voff = (unsigned)tid * (DIM * 2);  // byte stride of one W row

    for (int ko = 0; ko < DIM / KCH; ++ko) {
        const int k0 = ko * KCH;

        // ---- stage W chunk 256x128 bf16 via async global->LDS (ASYNCcnt) ----
        {
            const __bf16* wsrc = Wb + k0;   // saddr; voff selects the row
            #pragma unroll
            for (int c = 0; c < 16; ++c) {
                asm volatile(
                    "global_load_async_to_lds_b128 %0, %1, %2 offset:%3"
                    :: "v"(lds_woff), "v"(voff), "s"(wsrc), "n"(c * 16)
                    : "memory");
            }
        }
        // ---- stage x chunk: 64 tokens x 128 dims, fp32 -> bf16 ----
        {
            const float* src = X + (size_t)(t0 + xrow) * DIM + k0 + xseg * 32;
            __bf16* dst = Xs + xrow * LSTRIDE + xseg * 32;
            #pragma unroll
            for (int i = 0; i < 8; ++i) {
                v4f f = *(const v4f*)(src + i * 4);
                v4bf h;
                h[0] = (__bf16)f[0]; h[1] = (__bf16)f[1];
                h[2] = (__bf16)f[2]; h[3] = (__bf16)f[3];
                *(v4bf*)(dst + i * 4) = h;
            }
        }
        asm volatile("s_wait_asynccnt 0x0" ::: "memory");
        __syncthreads();

        // ---- 4 WMMA k-steps of K=32 ----
        #pragma unroll
        for (int ks = 0; ks < KCH / 32; ++ks) {
            const int sel  = lane >> 4;
            const int arow = tgroup * 16 + (lane & 15);
            // A layout: lanes 0-15 K{0..7,16..23}, lanes 16-31 K{8..15,24..31}
            const __bf16* ap = Xs + arow * LSTRIDE + ks * 32 + sel * 8;
            v8bf alo = *(const v8bf*)(ap);
            v8bf ahi = *(const v8bf*)(ap + 16);
            v16bf a = __builtin_shufflevector(alo, ahi,
                0,1,2,3,4,5,6,7,8,9,10,11,12,13,14,15);

            // Preload ALL B fragments first so ds_loads overlap the WMMAs
            v16bf bfr[8];
            #pragma unroll
            for (int n = 0; n < 8; ++n) {
                const int e = ehalf * 128 + n * 16 + (lane & 15);
                const __bf16* bp = Ws + e * LSTRIDE + ks * 32 + sel * 8;
                v8bf blo = *(const v8bf*)(bp);
                v8bf bhi = *(const v8bf*)(bp + 16);
                bfr[n] = __builtin_shufflevector(blo, bhi,
                    0,1,2,3,4,5,6,7,8,9,10,11,12,13,14,15);
            }
            #pragma unroll
            for (int n = 0; n < 8; ++n) {
                acc[n] = __builtin_amdgcn_wmma_f32_16x16x32_bf16(
                    false, a, false, bfr[n], (short)0, acc[n], false, false);
            }
        }
        __syncthreads();
    }

    // ---- scatter logits to LDS (C layout: VGPR j -> M=j or j+8 by lane half) ----
    {
        const int c   = lane & 15;
        const int mof = (lane >> 4) * 8;
        #pragma unroll
        for (int n = 0; n < 8; ++n) {
            const int e = ehalf * 128 + n * 16 + c;
            #pragma unroll
            for (int j = 0; j < 8; ++j) {
                const int tok = tgroup * 16 + mof + j;
                Lg[tok * LOGSTR + e] = acc[n][j];
            }
        }
        if (tid < NEXP) Bs[tid] = B[tid];
    }
    __syncthreads();

    // ---- per-token sigmoid + biased top-8 + normalize ----
    if (tid < MTILE) {
        float* row = Lg + tid * LOGSTR;
        for (int e = 0; e < NEXP; ++e) {
            const float v = row[e];
            row[e] = 1.0f / (1.0f + __expf(-v));
        }
        float wv[TOPK]; int wi[TOPK]; float sum = 0.f;
        for (int j = 0; j < TOPK; ++j) {
            float best = -3.4e38f; int bi = 0;
            for (int e = 0; e < NEXP; ++e) {
                const float s = row[e] + Bs[e];   // bias only affects selection
                if (s > best) { best = s; bi = e; }
            }
            const float w = row[bi];              // original sigmoid score
            row[bi] = -3.4e38f;
            wv[j] = w; wi[j] = bi; sum += w;
        }
        const float scale = 2.5f / (sum + 1e-20f);
        const int gt = t0 + tid;
        #pragma unroll
        for (int j = 0; j < TOPK; ++j) {
            outW[gt * TOPK + j] = wv[j] * scale;
            outI[gt * TOPK + j] = wi[j];
        }
    }
}

extern "C" void kernel_launch(void* const* d_in, const int* in_sizes, int n_in,
                              void* d_out, int out_size, void* d_ws, size_t ws_size,
                              hipStream_t stream) {
    const float* X = (const float*)d_in[0];   // [4,4096,2048] f32
    const float* W = (const float*)d_in[1];   // [256,2048]   f32
    const float* B = (const float*)d_in[2];   // [256]        f32

    const int tokens = in_sizes[0] / DIM;     // 16384
    float* outW = (float*)d_out;                        // [tokens,8] f32
    int*   outI = (int*)d_out + (size_t)tokens * TOPK;  // [tokens,8] i32 (raw bits)

    __bf16* Wb = (__bf16*)d_ws;               // 256*2048*2 = 1MB scratch

    // pre-convert W to bf16 (NEXP*DIM / (256 threads * 4 elems) blocks)
    cvt_w_kernel<<<(NEXP * DIM) / (256 * 4), 256, 0, stream>>>(W, Wb);

    const int blocks = tokens / MTILE;        // 256
    gate_topk_kernel<<<blocks, 256, 0, stream>>>(X, Wb, B, outW, outI);
}
